// MOELayer_80882824118674
// MI455X (gfx1250) — matrix operand order repaired and checked by reading, not compiled, not run on hardware
//
#include <hip/hip_runtime.h>
#include <stdint.h>

// ---------------------------------------------------------------------------
// Types for CDNA5 WMMA (wave32): v16bf A/B fragments, v8f accumulator
// ---------------------------------------------------------------------------
typedef __attribute__((ext_vector_type(16))) __bf16 v16bf;
typedef __attribute__((ext_vector_type(8)))  float  v8f;

union FragBF { uint4 u[2]; v16bf v; };

// pack two f32 -> packed bf16 pair (round-half-up) with one v_perm_b32
__device__ __forceinline__ unsigned pk2bf(float a, float b) {
  unsigned ua = __float_as_uint(a) + 0x8000u;
  unsigned ub = __float_as_uint(b) + 0x8000u;
  // result = { ub[3], ub[2], ua[3], ua[2] } bytes = (bf(b)<<16)|bf(a)
  return __builtin_amdgcn_perm(ub, ua, 0x07060302u);
}

__device__ __forceinline__ unsigned short f2bf(float f) {
  unsigned u = __float_as_uint(f) + 0x8000u;
  return (unsigned short)(u >> 16);
}

// ---------------------------------------------------------------------------
// 1) Gating: one wave per token. logits = x_row @ wg, softmax, top-1.
// ---------------------------------------------------------------------------
__global__ __launch_bounds__(256) void gate_kernel(
    const float* __restrict__ x, const float* __restrict__ wg,
    int* __restrict__ eidx, float* __restrict__ gate, int S, int m)
{
  int wave = threadIdx.x >> 5;
  int lane = threadIdx.x & 31;
  int token = blockIdx.x * 8 + wave;
  if (token >= S) return;
  const float* xr = x + (long long)token * m;
  float acc[8];
#pragma unroll
  for (int e = 0; e < 8; ++e) acc[e] = 0.f;
  for (int k = lane; k < m; k += 32) {
    float xv = xr[k];
    const float4* wr = (const float4*)(wg + (long long)k * 8);
    float4 w0 = wr[0], w1 = wr[1];
    acc[0] = fmaf(xv, w0.x, acc[0]);
    acc[1] = fmaf(xv, w0.y, acc[1]);
    acc[2] = fmaf(xv, w0.z, acc[2]);
    acc[3] = fmaf(xv, w0.w, acc[3]);
    acc[4] = fmaf(xv, w1.x, acc[4]);
    acc[5] = fmaf(xv, w1.y, acc[5]);
    acc[6] = fmaf(xv, w1.z, acc[6]);
    acc[7] = fmaf(xv, w1.w, acc[7]);
  }
#pragma unroll
  for (int off = 16; off > 0; off >>= 1) {
#pragma unroll
    for (int e = 0; e < 8; ++e) acc[e] += __shfl_xor(acc[e], off, 32);
  }
  if (lane == 0) {
    float mx = acc[0]; int best = 0;
#pragma unroll
    for (int e = 1; e < 8; ++e) if (acc[e] > mx) { mx = acc[e]; best = e; }
    float sum = 0.f;
#pragma unroll
    for (int e = 0; e < 8; ++e) sum += __expf(acc[e] - mx);
    eidx[token] = best;
    gate[token] = 1.0f / sum;    // softmax value at argmax
  }
}

// ---------------------------------------------------------------------------
// 2) Deterministic per-expert cumsum (token order), single wave32.
// ---------------------------------------------------------------------------
__global__ void scan_kernel(const int* __restrict__ eidx,
                            int* __restrict__ loc, int S)
{
  int lane = threadIdx.x;             // 0..31
  unsigned lt = (1u << lane) - 1u;
  int counts[8];
#pragma unroll
  for (int e = 0; e < 8; ++e) counts[e] = 0;
  for (int base = 0; base < S; base += 32) {
    int e = eidx[base + lane];
    int myloc = 0;
#pragma unroll
    for (int ee = 0; ee < 8; ++ee) {
      unsigned bal = (unsigned)__ballot(e == ee);   // wave32: low 32 bits
      if (ee == e) myloc = counts[ee] + __popc(bal & lt);
      counts[ee] += __popc(bal);                    // uniform across lanes
    }
    loc[base + lane] = myloc;
  }
}

// ---------------------------------------------------------------------------
// 3) Zero-fill (dispatch buffer), grid-stride uint4 stores
// ---------------------------------------------------------------------------
__global__ void zero_kernel(uint4* __restrict__ p, long long n4) {
  long long i = (long long)blockIdx.x * blockDim.x + threadIdx.x;
  long long stride = (long long)gridDim.x * blockDim.x;
  uint4 z; z.x = z.y = z.z = z.w = 0u;
  for (; i < n4; i += stride) p[i] = z;
}

// ---------------------------------------------------------------------------
// 4) Dispatch: scatter surviving tokens into [E][C][m], fp32 -> bf16
// ---------------------------------------------------------------------------
__global__ __launch_bounds__(256) void dispatch_kernel(
    const float* __restrict__ x, const int* __restrict__ eidx,
    const int* __restrict__ loc, unsigned short* __restrict__ disp,
    int m, int C)
{
  int token = blockIdx.x;
  int l = loc[token];
  if (l >= C) return;                       // dropped (over capacity)
  int e = eidx[token];
  unsigned short* dst = disp + ((long long)e * C + l) * m;
  const float* src = x + (long long)token * m;
  for (int c = threadIdx.x * 4; c < m; c += 256 * 4) {
    float4 v = *(const float4*)(src + c);
    uint2 pk;
    pk.x = pk2bf(v.x, v.y);
    pk.y = pk2bf(v.z, v.w);
    *(uint2*)(dst + c) = pk;
  }
}

// ---------------------------------------------------------------------------
// 5/6) Batched expert GEMM: D = A(bf16) x B(f32->bf16), f32 accumulate.
//   A tile: HBM -> LDS via global_load_async_to_lds_b128 (ASYNCcnt), no VGPRs.
//   B tile: column-wise b32 loads, v_perm pack, one ds_store_b128 per chunk
//           into the [N][K]-transposed LDS tile.
//   Double-buffered LDS (single __shared__ array => asm LDS offsets from 0);
//   one barrier per K-step; async + B loads for k+1 issued before WMMAs of k.
//   Block tile 128x128x32, 8 waves; wave = 2x4 grid of 16x16 WMMA tiles.
// ---------------------------------------------------------------------------
#define BM 128
#define BN 128
#define BK 32
#define LSTRIDE 40   // BK + 8 pad (row = 80B, multiple of 16B)

#define A_ELEMS (BM * LSTRIDE)          // per buffer
#define B_ELEMS (BN * LSTRIDE)

template<bool RELU, bool OUT_BF16>
__global__ __launch_bounds__(256, 1) void expert_gemm_kernel(
    const unsigned short* __restrict__ Abase,
    const float* __restrict__ Bbase,
    void* __restrict__ Obase,
    int N, int K,
    long long strideA, long long strideB, long long strideO)
{
  // single shared array: A buffers at LDS offset 0, B buffers after
  __shared__ unsigned short smem[2 * A_ELEMS + 2 * B_ELEMS];

  const int e = blockIdx.z;
  const unsigned short* A = Abase + (long long)e * strideA;
  const float* B = Bbase + (long long)e * strideB;

  const int tid = threadIdx.x;
  const int lane = tid & 31;
  const int wv = tid >> 5;
  const int l15 = lane & 15;
  const int hs = lane >> 4;            // half-wave select
  const int m_base = (wv & 3) * 32;    // wave's M offset in tile
  const int n_base = (wv >> 2) * 64;   // wave's N offset in tile
  const int blockM = blockIdx.y * BM;
  const int blockN = blockIdx.x * BN;
  const int a_k0 = hs * 8;             // A frag: K {0..7,16..23}/{8..15,24..31}
  const int b_k0 = hs * 8;             // B frag: K {0..7,16..23}/{8..15,24..31}

  // per-thread staging chunk coordinates (constant across k-steps)
  const int a_row0 = tid >> 2;
  const int a_ko0  = (tid & 3) * 8;
  const int a_row1 = (tid + 256) >> 2;
  const int a_ko1  = ((tid + 256) & 3) * 8;
  const int b_n0 = tid & 127;
  const int b_g0 = tid >> 7;           // kgroup 0/1
  const int b_g1 = b_g0 + 2;           // kgroup 2/3

  // LDS byte offsets for the async A stores (smem is at LDS offset 0)
  const unsigned ldsA0 = (unsigned)((a_row0 * LSTRIDE + a_ko0) * 2);
  const unsigned ldsA1 = (unsigned)((a_row1 * LSTRIDE + a_ko1) * 2);

  // running global pointers (advance by BK each k-step)
  const unsigned short* ap0 = A + (long long)(blockM + a_row0) * K + a_ko0;
  const unsigned short* ap1 = A + (long long)(blockM + a_row1) * K + a_ko1;
  const float* bp0 = B + (long long)(b_g0 * 8) * N + blockN + b_n0;
  const float* bp1 = B + (long long)(b_g1 * 8) * N + blockN + b_n0;

  float bf0[8], bf1[8];

  auto async_A = [&](const unsigned short* p0, const unsigned short* p1, int b) {
    unsigned l0 = ldsA0 + (unsigned)(b * (A_ELEMS * 2));
    unsigned l1 = ldsA1 + (unsigned)(b * (A_ELEMS * 2));
    asm volatile("global_load_async_to_lds_b128 %0, %1, off"
                 :: "v"(l0), "v"(p0) : "memory");
    asm volatile("global_load_async_to_lds_b128 %0, %1, off"
                 :: "v"(l1), "v"(p1) : "memory");
  };
  auto load_B = [&](const float* p0, const float* p1) {
#pragma unroll
    for (int j = 0; j < 8; ++j) bf0[j] = p0[(long long)j * N];
#pragma unroll
    for (int j = 0; j < 8; ++j) bf1[j] = p1[(long long)j * N];
  };
  auto store_B = [&](int b) {
    unsigned short* Bb = smem + 2 * A_ELEMS + b * B_ELEMS;
    uint4 w0, w1;
    w0.x = pk2bf(bf0[0], bf0[1]);
    w0.y = pk2bf(bf0[2], bf0[3]);
    w0.z = pk2bf(bf0[4], bf0[5]);
    w0.w = pk2bf(bf0[6], bf0[7]);
    w1.x = pk2bf(bf1[0], bf1[1]);
    w1.y = pk2bf(bf1[2], bf1[3]);
    w1.z = pk2bf(bf1[4], bf1[5]);
    w1.w = pk2bf(bf1[6], bf1[7]);
    *(uint4*)(&Bb[b_n0 * LSTRIDE + b_g0 * 8]) = w0;
    *(uint4*)(&Bb[b_n0 * LSTRIDE + b_g1 * 8]) = w1;
  };

  v8f zero = {0.f, 0.f, 0.f, 0.f, 0.f, 0.f, 0.f, 0.f};
  v8f acc[2][4];
#pragma unroll
  for (int mt = 0; mt < 2; ++mt)
#pragma unroll
    for (int nt = 0; nt < 4; ++nt) acc[mt][nt] = zero;

  // ---- prologue: stage k0 = 0 into buffer 0 ----
  async_A(ap0, ap1, 0);
  load_B(bp0, bp1);
  store_B(0);
  asm volatile("s_wait_asynccnt 0x0" ::: "memory");
  __syncthreads();

  int buf = 0;
  for (int k0 = 0; k0 < K; k0 += BK) {
    const bool more = (k0 + BK) < K;
    if (more) {
      ap0 += BK; ap1 += BK;
      bp0 += (long long)BK * N; bp1 += (long long)BK * N;
      async_A(ap0, ap1, buf ^ 1);     // HBM->LDS, no VGPRs, ASYNCcnt
      load_B(bp0, bp1);               // raw f32, waited only at store_B
    }

    // --- fragments from LDS buffer `buf` ---
    const unsigned short* Ab = smem + buf * A_ELEMS;
    const unsigned short* Bb = smem + 2 * A_ELEMS + buf * B_ELEMS;
    FragBF af[2];
#pragma unroll
    for (int mt = 0; mt < 2; ++mt) {
      const unsigned short* p = &Ab[(m_base + mt * 16 + l15) * LSTRIDE + a_k0];
      af[mt].u[0] = *(const uint4*)p;           // K = a_k0 .. a_k0+7
      af[mt].u[1] = *(const uint4*)(p + 16);    // K = a_k0+16 .. a_k0+23
    }
    FragBF bfb[4];
#pragma unroll
    for (int nt = 0; nt < 4; ++nt) {
      const unsigned short* p = &Bb[(n_base + nt * 16 + l15) * LSTRIDE + b_k0];
      bfb[nt].u[0] = *(const uint4*)p;          // K = b_k0 .. b_k0+7
      bfb[nt].u[1] = *(const uint4*)(p + 16);   // K = b_k0+16 .. b_k0+23
    }
    // --- 8 WMMAs per wave per K-step ---
#pragma unroll
    for (int mt = 0; mt < 2; ++mt)
#pragma unroll
      for (int nt = 0; nt < 4; ++nt)
        acc[mt][nt] = __builtin_amdgcn_wmma_f32_16x16x32_bf16(
            false, af[mt].v, false, bfb[nt].v, (short)0, acc[mt][nt],
            false, false);

    if (more) {
      store_B(buf ^ 1);               // pack + ds_store_b128 (after WMMAs)
      asm volatile("s_wait_asynccnt 0x0" ::: "memory");
    }
    __syncthreads();
    buf ^= 1;
  }

  // --- epilogue: C/D layout — VGPR i holds (M = i + hs*8, N = l15) ---
#pragma unroll
  for (int mt = 0; mt < 2; ++mt) {
#pragma unroll
    for (int nt = 0; nt < 4; ++nt) {
      long long row0 = blockM + m_base + mt * 16 + hs * 8;
      long long col = blockN + n_base + nt * 16 + l15;
#pragma unroll
      for (int i = 0; i < 8; ++i) {
        float v = acc[mt][nt][i];
        if (RELU) v = fmaxf(v, 0.0f);
        long long off = (long long)e * strideO + (row0 + i) * (long long)N + col;
        if (OUT_BF16) ((unsigned short*)Obase)[off] = f2bf(v);
        else          ((float*)Obase)[off] = v;
      }
    }
  }
}

// ---------------------------------------------------------------------------
// 7) Combine: out[token] = gate * eo[e, loc] (zeros for dropped tokens)
// ---------------------------------------------------------------------------
__global__ __launch_bounds__(256) void combine_kernel(
    const float* __restrict__ eo, const int* __restrict__ eidx,
    const int* __restrict__ loc, const float* __restrict__ gate,
    float* __restrict__ out, int m, int C)
{
  int token = blockIdx.x;
  int l = loc[token];
  int e = eidx[token];
  float g = gate[token];
  bool keep = (l < C);
  const float* src = eo + ((long long)e * C + (keep ? l : 0)) * m;
  for (int c = threadIdx.x * 4; c < m; c += 256 * 4) {
    float4 v;
    if (keep) {
      v = *(const float4*)(src + c);
      v.x *= g; v.y *= g; v.z *= g; v.w *= g;
    } else {
      v.x = v.y = v.z = v.w = 0.0f;
    }
    *(float4*)(out + (long long)token * m + c) = v;
  }
}

// ---------------------------------------------------------------------------
// Launcher
// ---------------------------------------------------------------------------
extern "C" void kernel_launch(void* const* d_in, const int* in_sizes, int n_in,
                              void* d_out, int out_size, void* d_ws, size_t ws_size,
                              hipStream_t stream)
{
  (void)n_in; (void)out_size; (void)ws_size;
  const float* x  = (const float*)d_in[0];   // [s, b, m] -> x2d [S, m]
  const float* wg = (const float*)d_in[1];   // [m, E]
  const float* w1 = (const float*)d_in[2];   // [E, m, ffn]
  const float* w2 = (const float*)d_in[3];   // [E, ffn, m]
  float* out = (float*)d_out;

  const int m = 1024, E = 8, ffn = 4096;
  const int S = in_sizes[0] / m;   // 8192 tokens
  const int C = S / E;             // 1024 capacity

  char* ws = (char*)d_ws;
  size_t off = 0;
  unsigned short* disp = (unsigned short*)(ws + off); off += (size_t)E * C * m   * 2; // 16MB bf16
  unsigned short* h    = (unsigned short*)(ws + off); off += (size_t)E * C * ffn * 2; // 64MB bf16
  float* eo            = (float*)(ws + off);          off += (size_t)E * C * m   * 4; // 32MB f32
  int* eidx            = (int*)(ws + off);            off += (size_t)S * 4;
  int* loc             = (int*)(ws + off);            off += (size_t)S * 4;
  float* gate          = (float*)(ws + off);          off += (size_t)S * 4;

  gate_kernel<<<S / 8, 256, 0, stream>>>(x, wg, eidx, gate, S, m);
  scan_kernel<<<1, 32, 0, stream>>>(eidx, loc, S);
  {
    long long n4 = (long long)E * C * m * 2 / 16;
    zero_kernel<<<4096, 256, 0, stream>>>((uint4*)disp, n4);
  }
  dispatch_kernel<<<S, 256, 0, stream>>>(x, eidx, loc, disp, m, C);

  // GEMM1: h = relu(disp @ w1)   (M=C, N=ffn, K=m), bf16 output
  expert_gemm_kernel<true, true><<<dim3(ffn / BN, C / BM, E), 256, 0, stream>>>(
      disp, w1, (void*)h, ffn, m,
      (long long)C * m, (long long)m * ffn, (long long)C * ffn);

  // GEMM2: eo = h @ w2           (M=C, N=m, K=ffn), f32 output
  expert_gemm_kernel<false, false><<<dim3(m / BN, C / BM, E), 256, 0, stream>>>(
      h, w2, (void*)eo, m, ffn,
      (long long)C * ffn, (long long)ffn * m, (long long)C * m);

  combine_kernel<<<S, 256, 0, stream>>>(eo, eidx, loc, gate, out, m, C);
}
